// HashingMemory_65352222375970
// MI455X (gfx1250) — matrix operand order repaired and checked by reading, not compiled, not run on hardware
//
#include <hip/hip_runtime.h>
#include <hip/hip_bf16.h>
#include <math.h>

// ---------------- problem constants ----------------
#define NTOK   2048          // B*T = 2*1024
#define CDIM   1024          // d_in
#define HK     2048          // HEADS*K_DIM
#define HALF   256           // K_DIM/2
#define NKEYS  512
#define KNN    32
#define VDIM   512
#define DOUT   1024
#define NHH    8             // HEADS*2 (head-half pairs)

typedef __bf16 v16bf __attribute__((ext_vector_type(16)));
typedef __bf16 v8bf  __attribute__((ext_vector_type(8)));
typedef float  v8f   __attribute__((ext_vector_type(8)));

static __device__ __forceinline__ unsigned short f32_to_bf16(float f) {
    unsigned int u = __float_as_uint(f);
    unsigned int r = u + 0x7FFFu + ((u >> 16) & 1u);   // round-to-nearest-even
    return (unsigned short)(r >> 16);
}

// ---------------- fp32 -> bf16 conversions ----------------
__global__ void conv_bf16(const float* __restrict__ src, unsigned short* __restrict__ dst, int n) {
    int i = blockIdx.x * blockDim.x + threadIdx.x;
    if (i < n) dst[i] = f32_to_bf16(src[i]);
}

// src is R x C row-major fp32; dst is C x R row-major bf16 (i.e. src^T)
__global__ void conv_t_bf16(const float* __restrict__ src, unsigned short* __restrict__ dst, int R, int C) {
    int c = blockIdx.x * 16 + threadIdx.x;
    int r = blockIdx.y * 16 + threadIdx.y;
    if (r < R && c < C) dst[(size_t)c * R + r] = f32_to_bf16(src[(size_t)r * C + c]);
}

// ---------------- LDS-staged, double-buffered WMMA bf16 GEMM ----------------
//   C[M,N] = A[M,K] * BT[N,K]^T (+bias, epilogue)
// epi: 0 = f32 out (+bias), 1 = bf16 out (+bias), 2 = f32 out with silu(x+bias)
// grid = (N/64, M/64); block = 256 threads = 8 waves.
// Per k-step: 64x32 A tile + 64x32 B tile staged into LDS[buf] with
// global_load_async_to_lds_b128; step i prefetches step i+1's tiles into the
// other buffer, then s_wait_asynccnt 2 (in-order completion => current buffer
// landed) so the global fetch overlaps the WMMA work of the current step.
// Wave w -> n-tile (w&3), m-tiles 2*(w>>2)+{0,1}: one B fragment feeds 2 WMMAs.
__global__ __launch_bounds__(256) void wmma_gemm_bf16_lds(
        const unsigned short* __restrict__ A, int lda,
        const unsigned short* __restrict__ BT, int ldb,
        const float* __restrict__ bias,
        float* __restrict__ outF, unsigned short* __restrict__ outB,
        int ldc, int K, int epi) {
    __shared__ unsigned short shA[2][64 * 32];   // 2 x 4 KB
    __shared__ unsigned short shB[2][64 * 32];   // 2 x 4 KB

    const int tid  = threadIdx.x;
    const int lane = tid & 31;
    const int wave = tid >> 5;
    const int r16  = lane & 15;
    const int hi   = lane >> 4;
    const int nt   = wave & 3;          // local n-tile 0..3
    const int mt   = (wave >> 2) * 2;   // local m-tile base: 0 or 2

    const int m0 = blockIdx.y * 64;
    const int n0 = blockIdx.x * 64;

    // cooperative tile copy: 256 threads x 16B = one 64x32 bf16 tile
    const int crow = tid >> 2;          // 0..63
    const int ccol = (tid & 3) * 8;     // 0,8,16,24 (elements)

    const unsigned short* ag = A  + (size_t)(m0 + crow) * lda + ccol;
    const unsigned short* bg = BT + (size_t)(n0 + crow) * ldb + ccol;
    const unsigned int ldsA[2] = {
        (unsigned int)(uintptr_t)&shA[0][crow * 32 + ccol],
        (unsigned int)(uintptr_t)&shA[1][crow * 32 + ccol] };
    const unsigned int ldsB[2] = {
        (unsigned int)(uintptr_t)&shB[0][crow * 32 + ccol],
        (unsigned int)(uintptr_t)&shB[1][crow * 32 + ccol] };

    auto issue = [&](int buf, int kk) {
        unsigned long long ga = (unsigned long long)(uintptr_t)(ag + kk);
        unsigned long long gb = (unsigned long long)(uintptr_t)(bg + kk);
        asm volatile(
            "global_load_async_to_lds_b128 %0, %2, off\n\t"
            "global_load_async_to_lds_b128 %1, %3, off"
            :
            : "v"(ldsA[buf]), "v"(ldsB[buf]), "v"(ga), "v"(gb)
            : "memory");
    };

    v8f acc0 = {};
    v8f acc1 = {};

    const int nsteps = K >> 5;
    issue(0, 0);                                 // prologue: first tiles
    for (int i = 0; i < nsteps; ++i) {
        const int p = i & 1;
        if (i + 1 < nsteps) {
            issue(1 - p, (i + 1) * 32);          // prefetch next tiles
            asm volatile("s_wait_asynccnt 0x2" ::: "memory");  // current landed
        } else {
            asm volatile("s_wait_asynccnt 0x0" ::: "memory");
        }
        __syncthreads();                         // tiles visible to all waves

        union { v16bf v; v8bf h[2]; } a0, a1, b;
        const unsigned short* sa0 = &shA[p][(mt * 16 + r16) * 32];
        const unsigned short* sa1 = sa0 + 16 * 32;
        const unsigned short* sb  = &shB[p][(nt * 16 + r16) * 32];
        // A 16x32 bf16 layout: lanes<16 hold K = 0..7 / 16..23, lanes>=16 +8
        a0.h[0] = *(const v8bf*)(sa0 + hi * 8);
        a0.h[1] = *(const v8bf*)(sa0 + hi * 8 + 16);
        a1.h[0] = *(const v8bf*)(sa1 + hi * 8);
        a1.h[1] = *(const v8bf*)(sa1 + hi * 8 + 16);
        // B 32x16 bf16 layout: lanes<16 hold K = 0..7 / 16..23, lanes>=16 +8
        b.h[0]  = *(const v8bf*)(sb + hi * 8);
        b.h[1]  = *(const v8bf*)(sb + hi * 8 + 16);

        acc0 = __builtin_amdgcn_wmma_f32_16x16x32_bf16(
            false, a0.v, false, b.v, (short)0, acc0, false, false);
        acc1 = __builtin_amdgcn_wmma_f32_16x16x32_bf16(
            false, a1.v, false, b.v, (short)0, acc1, false, false);

        __syncthreads();                         // readers done before refill
    }

    const int n = n0 + nt * 16 + r16;
    const float bn = bias ? bias[n] : 0.0f;
#pragma unroll
    for (int t = 0; t < 2; ++t) {
        v8f acc;
        if (t == 0) acc = acc0; else acc = acc1;
#pragma unroll
        for (int r = 0; r < 8; ++r) {
            int m = m0 + (mt + t) * 16 + r + hi * 8;
            float v = acc[r] + bn;
            if (epi == 2) v = v * __builtin_amdgcn_rcpf(1.0f + __expf(-v));  // silu
            if (epi == 1) outB[(size_t)m * ldc + n] = f32_to_bf16(v);
            else          outF[(size_t)m * ldc + n] = v;
        }
    }
}

// ---------------- top-32 of 512 per (token, head-half) ----------------
__global__ void topk_half(const float* __restrict__ scores,
                          float* __restrict__ sTop, int* __restrict__ iTop) {
    int u = blockIdx.x * blockDim.x + threadIdx.x;
    if (u >= NTOK * NHH) return;
    int n = u >> 3, hh = u & 7;
    const float* p = scores + (size_t)n * (NHH * NKEYS) + (size_t)hh * NKEYS;

    float tv[KNN]; int ti[KNN];
#pragma unroll
    for (int i = 0; i < KNN; ++i) { tv[i] = -1e30f; ti[i] = 0; }
    for (int k = 0; k < NKEYS; ++k) {
        float s = p[k];
        if (s > tv[KNN - 1]) {
            int pos = KNN - 1;
            while (pos > 0 && tv[pos - 1] < s) {
                tv[pos] = tv[pos - 1]; ti[pos] = ti[pos - 1]; --pos;
            }
            tv[pos] = s; ti[pos] = k;
        }
    }
    float* so = sTop + (size_t)u * KNN;
    int*   io = iTop + (size_t)u * KNN;
    for (int i = 0; i < KNN; ++i) { so[i] = tv[i]; io[i] = ti[i]; }
}

// ---------------- combine halves: top-32 of 32x32 sums, softmax, value idx -----------
__global__ void topk_combine(const float* __restrict__ sTop, const int* __restrict__ iTop,
                             float* __restrict__ wOut, int* __restrict__ idxOut) {
    int u = blockIdx.x * blockDim.x + threadIdx.x;
    if (u >= NTOK * 4) return;
    int n = u >> 2, h = u & 3;
    const float* s1 = sTop + (size_t)(n * NHH + h * 2 + 0) * KNN;
    const float* s2 = sTop + (size_t)(n * NHH + h * 2 + 1) * KNN;
    const int*   i1 = iTop + (size_t)(n * NHH + h * 2 + 0) * KNN;
    const int*   i2 = iTop + (size_t)(n * NHH + h * 2 + 1) * KNN;

    float tv[KNN]; int tc[KNN];
#pragma unroll
    for (int i = 0; i < KNN; ++i) { tv[i] = -1e30f; tc[i] = 0; }
    for (int a = 0; a < KNN; ++a) {
        float sa = s1[a];
        for (int b = 0; b < KNN; ++b) {
            float s = sa + s2[b];
            if (s > tv[KNN - 1]) {
                int pos = KNN - 1;
                while (pos > 0 && tv[pos - 1] < s) {
                    tv[pos] = tv[pos - 1]; tc[pos] = tc[pos - 1]; --pos;
                }
                tv[pos] = s; tc[pos] = a * KNN + b;
            }
        }
    }
    // softmax over the 32 best (tv[0] is the max: sorted descending)
    float mx = tv[0], sum = 0.0f, e[KNN];
    for (int i = 0; i < KNN; ++i) { e[i] = __expf(tv[i] - mx); sum += e[i]; }
    float inv = 1.0f / sum;
    float* wo = wOut + (size_t)n * (4 * KNN) + h * KNN;
    int*   io = idxOut + (size_t)n * (4 * KNN) + h * KNN;
    for (int i = 0; i < KNN; ++i) {
        wo[i] = e[i] * inv;
        io[i] = i1[tc[i] >> 5] * NKEYS + i2[tc[i] & 31];
    }
}

// ---------------- weighted value gather + gate multiply -> bf16 ----------------
// one block per token; 512 threads (one per value dim); coalesced across d
__global__ void gather_weighted(const float* __restrict__ w, const int* __restrict__ idx,
                                const float* __restrict__ values, const float* __restrict__ gate,
                                unsigned short* __restrict__ gvb) {
    __shared__ float sw[128];
    __shared__ int   si[128];
    int n = blockIdx.x;
    int d = threadIdx.x;
    if (threadIdx.x < 128) {
        sw[threadIdx.x] = w[(size_t)n * 128 + threadIdx.x];
        si[threadIdx.x] = idx[(size_t)n * 128 + threadIdx.x];
    }
    __syncthreads();
    float acc = 0.0f;
    for (int j = 0; j < 128; ++j) {
        if (j + 4 < 128)
            __builtin_prefetch(&values[(size_t)si[j + 4] * VDIM + d], 0, 0);
        acc += sw[j] * values[(size_t)si[j] * VDIM + d];
    }
    float g = gate[(size_t)n * VDIM + d];
    gvb[(size_t)n * VDIM + d] = f32_to_bf16(g * acc);
}

// ---------------- host launch ----------------
extern "C" void kernel_launch(void* const* d_in, const int* in_sizes, int n_in,
                              void* d_out, int out_size, void* d_ws, size_t ws_size,
                              hipStream_t stream) {
    (void)in_sizes; (void)n_in; (void)out_size; (void)ws_size;
    const float* x      = (const float*)d_in[0];
    const float* Wq     = (const float*)d_in[1];
    const float* bq     = (const float*)d_in[2];
    const float* keys   = (const float*)d_in[3];
    const float* values = (const float*)d_in[4];
    const float* Ws     = (const float*)d_in[5];
    const float* bs     = (const float*)d_in[6];
    const float* Wv     = (const float*)d_in[7];
    const float* bv     = (const float*)d_in[8];
    float* out = (float*)d_out;

    char* ws = (char*)d_ws;
    size_t off = 0;
    auto alloc = [&](size_t bytes) -> void* {
        void* p = ws + off;
        off += (bytes + 255) & ~(size_t)255;
        return p;
    };
    unsigned short* xbf    = (unsigned short*)alloc((size_t)NTOK * CDIM * 2);
    unsigned short* WqT    = (unsigned short*)alloc((size_t)HK * CDIM * 2);
    unsigned short* keysbf = (unsigned short*)alloc((size_t)NHH * NKEYS * HALF * 2);
    unsigned short* WsT    = (unsigned short*)alloc((size_t)VDIM * CDIM * 2);
    unsigned short* WvT    = (unsigned short*)alloc((size_t)DOUT * VDIM * 2);
    unsigned short* qbf    = (unsigned short*)alloc((size_t)NTOK * HK * 2);
    float*          scores = (float*)alloc((size_t)NTOK * NHH * NKEYS * 4);
    float*          sTop   = (float*)alloc((size_t)NTOK * NHH * KNN * 4);
    int*            iTop   = (int*)alloc((size_t)NTOK * NHH * KNN * 4);
    float*          wBuf   = (float*)alloc((size_t)NTOK * 4 * KNN * 4);
    int*            idxBuf = (int*)alloc((size_t)NTOK * 4 * KNN * 4);
    float*          gateF  = (float*)alloc((size_t)NTOK * VDIM * 4);
    unsigned short* gvbf   = (unsigned short*)alloc((size_t)NTOK * VDIM * 2);

    // --- conversions / transposes ---
    conv_bf16<<<(NTOK * CDIM) / 256, 256, 0, stream>>>(x, xbf, NTOK * CDIM);
    conv_bf16<<<(NHH * NKEYS * HALF) / 256, 256, 0, stream>>>(keys, keysbf, NHH * NKEYS * HALF);
    conv_t_bf16<<<dim3(HK / 16, CDIM / 16), dim3(16, 16), 0, stream>>>(Wq, WqT, CDIM, HK);
    conv_t_bf16<<<dim3(VDIM / 16, CDIM / 16), dim3(16, 16), 0, stream>>>(Ws, WsT, CDIM, VDIM);
    conv_t_bf16<<<dim3(DOUT / 16, VDIM / 16), dim3(16, 16), 0, stream>>>(Wv, WvT, VDIM, DOUT);

    // --- q = x @ Wq + bq  (bf16 out for the score GEMMs) ---
    wmma_gemm_bf16_lds<<<dim3(HK / 64, NTOK / 64), 256, 0, stream>>>(
        xbf, CDIM, WqT, CDIM, bq, nullptr, qbf, HK, CDIM, /*epi=*/1);

    // --- scores[n, hh, k] = q[n, hh*256:...] . keys[hh, k, :] ---
    for (int hh = 0; hh < NHH; ++hh) {
        wmma_gemm_bf16_lds<<<dim3(NKEYS / 64, NTOK / 64), 256, 0, stream>>>(
            qbf + (size_t)hh * HALF, HK,
            keysbf + (size_t)hh * NKEYS * HALF, HALF,
            nullptr, scores + (size_t)hh * NKEYS, nullptr,
            NHH * NKEYS, HALF, /*epi=*/0);
    }

    // --- top-k selection + combine + softmax ---
    topk_half<<<(NTOK * NHH) / 256, 256, 0, stream>>>(scores, sTop, iTop);
    topk_combine<<<(NTOK * 4) / 256, 256, 0, stream>>>(sTop, iTop, wBuf, idxBuf);

    // --- gate = silu(x @ Ws + bs) ---
    wmma_gemm_bf16_lds<<<dim3(VDIM / 64, NTOK / 64), 256, 0, stream>>>(
        xbf, CDIM, WsT, CDIM, bs, gateF, nullptr, VDIM, CDIM, /*epi=*/2);

    // --- gv = gate * (weighted value gather), bf16 ---
    gather_weighted<<<NTOK, VDIM, 0, stream>>>(wBuf, idxBuf, values, gateF, gvbf);

    // --- out = gv @ Wv + bv ---
    wmma_gemm_bf16_lds<<<dim3(DOUT / 64, NTOK / 64), 256, 0, stream>>>(
        gvbf, VDIM, WvT, VDIM, bv, out, nullptr, DOUT, VDIM, /*epi=*/0);
}
